// GPT2Attention_16295105921660
// MI455X (gfx1250) — compile-verified, matrix operand order checked by reading, not run on hardware
//
#include <hip/hip_runtime.h>

// ---------------------------------------------------------------------------
// GPT-2 attention + LMF value fusion for MI455X (gfx1250, wave32, WMMA).
// Heavy math via v_wmma_f32_16x16x32_f16; tiles streamed with
// global_load_async_to_lds_b128 (ASYNCcnt) into double-buffered LDS.
// All fragment gathers are contiguous-K -> 2x ds_load_b128 per fragment
// (weights pre-transposed, fused V stored head-transposed).
// ---------------------------------------------------------------------------

typedef _Float16 v16h __attribute__((ext_vector_type(16)));
typedef float    v8f  __attribute__((ext_vector_type(8)));

#define BB 4
#define SS 2048
#define DD 1024
#define HH 16
#define HDI 64
#define RR 4
#define LLAT 256

static __device__ __forceinline__ v8f wmma16(v16h a, v16h b, v8f c) {
  // (neg_a, A, neg_b, B, c_mod, C, reuse_a, reuse_b)
  return __builtin_amdgcn_wmma_f32_16x16x32_f16(false, a, false, b, (short)0, c,
                                                false, false);
}

// K-index pattern for 16-bit A/B fragments (ISA 7.12.2): lane half h, elem e.
// Contiguous in chunks of 8 -> vectorizes to ds_load_b128.
static __device__ __forceinline__ int kpat(int e, int h) {
  return e + 8 * h + ((e & 8) ? 8 : 0);
}

static __device__ __forceinline__ v8f zero8() {
  v8f z;
#pragma unroll
  for (int i = 0; i < 8; ++i) z[i] = 0.0f;
  return z;
}

// Async global->LDS 128-bit copy (CDNA5 async path, ASYNCcnt tracked).
static __device__ __forceinline__ void async_b128(unsigned int lds_off,
                                                  const void* gaddr) {
  asm volatile("global_load_async_to_lds_b128 %0, %1, off"
               :
               : "v"(lds_off), "v"(gaddr)
               : "memory");
}
static __device__ __forceinline__ void wait_async0() {
  asm volatile("s_wait_asynccnt 0" ::: "memory");
}
static __device__ __forceinline__ unsigned int lds_off(const void* p) {
  return (unsigned int)(unsigned long long)p;  // low 32b of flat = LDS offset
}

// ---------------------------------------------------------------------------
// Elementwise f32 -> f16 conversion (activations)
// ---------------------------------------------------------------------------
__global__ void k_cvt_f32_f16(const float* __restrict__ in,
                              _Float16* __restrict__ out, int n) {
  int i = blockIdx.x * blockDim.x + threadIdx.x;
  if (i < n) out[i] = (_Float16)in[i];
}

// ---------------------------------------------------------------------------
// LDS-tiled transpose + f32 -> f16:  out[N][K] = (f16) in[K][N]
// Grid (N/32, K/32), 256 threads (32x8).
// ---------------------------------------------------------------------------
__global__ __launch_bounds__(256) void k_cvt_t(const float* __restrict__ in,
                                               _Float16* __restrict__ out,
                                               int K, int N) {
  __shared__ float tile[32][33];
  const int kb = blockIdx.y * 32, nb = blockIdx.x * 32;
  const int tx = threadIdx.x & 31, ty = threadIdx.x >> 5;
#pragma unroll
  for (int i = ty; i < 32; i += 8)
    tile[i][tx] = in[(size_t)(kb + i) * N + nb + tx];
  __syncthreads();
#pragma unroll
  for (int i = ty; i < 32; i += 8)
    out[(size_t)(nb + i) * K + kb + tx] = (_Float16)tile[tx][i];
}

// ---------------------------------------------------------------------------
// Sum text_w over rank R; rows 0..1023 -> TRANSPOSED f16 weight [o][i],
// row 1024 (+text_b sum) -> f32 bias vector.
// ---------------------------------------------------------------------------
__global__ void k_prep_text(const float* __restrict__ tw,
                            const float* __restrict__ tb,
                            _Float16* __restrict__ w_ht,
                            float* __restrict__ bias_out) {
  int idx = blockIdx.x * blockDim.x + threadIdx.x;
  if (idx >= 1025 * 1024) return;
  int i = idx >> 10, o = idx & 1023;
  float s = 0.f;
#pragma unroll
  for (int r = 0; r < RR; ++r) s += tw[((size_t)r * 1025 + i) * 1024 + o];
  if (i < 1024) {
    w_ht[(size_t)o * 1024 + i] = (_Float16)s;  // transposed store
  } else {
    float b = 0.f;
#pragma unroll
    for (int r = 0; r < RR; ++r) b += tb[r * 1024 + o];
    bias_out[o] = s + b;
  }
}

// ---------------------------------------------------------------------------
// latent_fusion[b][o] = sum_r ( latent1[b] . lat_w[r][:,o] ) + sum_r lat_b[r][o]
// ---------------------------------------------------------------------------
__global__ void k_latfus(const float* __restrict__ latent,
                         const float* __restrict__ lw,
                         const float* __restrict__ lb,
                         float* __restrict__ out) {
  int t = blockIdx.x * blockDim.x + threadIdx.x;
  if (t >= BB * DD) return;
  int b = t >> 10, o = t & 1023;
  float acc = 0.f;
  for (int r = 0; r < RR; ++r) {
    const float* w = lw + ((size_t)r * (LLAT + 1)) * DD + o;
    float s = w[(size_t)LLAT * DD];  // bias-1 row
    for (int i = 0; i < LLAT; ++i) s += latent[b * LLAT + i] * w[(size_t)i * DD];
    acc += s + lb[r * DD + o];
  }
  out[t] = acc;
}

// ---------------------------------------------------------------------------
// Tiled f16 WMMA GEMM:  C[M,N] = A[M,K] @ Bt[N,K]^T + bias[N]
// (B passed TRANSPOSED so both A and B fragment gathers are contiguous-K.)
// Block tile 128x128, 8 waves (4x2), wave tile 32x64 (2x4 WMMA frags), BK=32.
// A/Bt tiles streamed with async b128 loads into double-buffered LDS.
// mode 0: out_f32[m*N+n] = c                          (final projection)
// mode 1: QKV split -> Qs (scaled 0.125), Ks (split-head), Vt (row-major)
// mode 2: c *= latf[b*1024+n]; -> out_h head-TRANSPOSED [b,h][hd][s]
// ---------------------------------------------------------------------------
#define TPAD 40  // tile row stride (f16): 80B, 16B-aligned chunks, 20-bank stagger

__global__ __launch_bounds__(256) void k_gemm_f16(
    const _Float16* __restrict__ A, const _Float16* __restrict__ Bt,
    const float* __restrict__ bias, int M, int N, int K, int mode,
    float* __restrict__ out_f32, _Float16* __restrict__ out_h,
    _Float16* __restrict__ qs, _Float16* __restrict__ ksp,
    _Float16* __restrict__ vt, const float* __restrict__ latf) {
  __shared__ __align__(16) _Float16 Alds[2][128 * TPAD];
  __shared__ __align__(16) _Float16 Blds[2][128 * TPAD];

  const int t = threadIdx.x;
  const int lane = t & 31, wave = t >> 5;
  const int wm = wave & 3, wn = wave >> 2;           // 4x2 wave grid
  const int mbase = blockIdx.y * 128, nbase = blockIdx.x * 128;
  const int hl = lane >> 4, n15 = lane & 15;

  // per-thread async-load slot (same shape for A and Bt: 128 rows x 32 K)
  const int lrow = t >> 1, lcol = (t & 1) * 16;  // 16 f16 = 2x b128 per tile

  v8f acc[2][4];
#pragma unroll
  for (int i = 0; i < 2; ++i)
#pragma unroll
    for (int j = 0; j < 4; ++j) acc[i][j] = zero8();

  const int nk = K >> 5;

  // prologue: stream tile 0 into buffer 0
  {
    const _Float16* ga = A + (size_t)(mbase + lrow) * K + lcol;
    unsigned int la = lds_off(&Alds[0][lrow * TPAD + lcol]);
    async_b128(la, ga);
    async_b128(la + 16, ga + 8);
    const _Float16* gb = Bt + (size_t)(nbase + lrow) * K + lcol;
    unsigned int lb = lds_off(&Blds[0][lrow * TPAD + lcol]);
    async_b128(lb, gb);
    async_b128(lb + 16, gb + 8);
  }

  for (int ki = 0; ki < nk; ++ki) {
    const int cur = ki & 1;
    wait_async0();    // my async loads for buf[cur] landed
    __syncthreads();  // everyone's landed; everyone done with buf[cur^1]

    if (ki + 1 < nk) {  // prefetch next K-tile into the other buffer
      const int kb = (ki + 1) << 5;
      const _Float16* ga = A + (size_t)(mbase + lrow) * K + kb + lcol;
      unsigned int la = lds_off(&Alds[cur ^ 1][lrow * TPAD + lcol]);
      async_b128(la, ga);
      async_b128(la + 16, ga + 8);
      const _Float16* gb = Bt + (size_t)(nbase + lrow) * K + kb + lcol;
      unsigned int lb = lds_off(&Blds[cur ^ 1][lrow * TPAD + lcol]);
      async_b128(lb, gb);
      async_b128(lb + 16, gb + 8);
    }

    const _Float16* Ac = Alds[cur];
    const _Float16* Bc = Blds[cur];
    v16h af[2], bf[4];
#pragma unroll
    for (int i = 0; i < 2; ++i) {
      int rbase = wm * 32 + i * 16 + n15;
#pragma unroll
      for (int e = 0; e < 16; ++e) af[i][e] = Ac[rbase * TPAD + kpat(e, hl)];
    }
#pragma unroll
    for (int j = 0; j < 4; ++j) {
      int cbase = wn * 64 + j * 16 + n15;
#pragma unroll
      for (int e = 0; e < 16; ++e) bf[j][e] = Bc[cbase * TPAD + kpat(e, hl)];
    }
#pragma unroll
    for (int i = 0; i < 2; ++i)
#pragma unroll
      for (int j = 0; j < 4; ++j) acc[i][j] = wmma16(af[i], bf[j], acc[i][j]);
  }

  // Epilogue. C layout: lane -> n = lane&15, row m = r + 8*(lane>=16).
#pragma unroll
  for (int i = 0; i < 2; ++i) {
#pragma unroll
    for (int j = 0; j < 4; ++j) {
#pragma unroll
      for (int r = 0; r < 8; ++r) {
        int gm = mbase + wm * 32 + i * 16 + r + 8 * hl;
        int gn = nbase + wn * 64 + j * 16 + n15;
        float c = acc[i][j][r] + bias[gn];
        if (mode == 0) {
          out_f32[(size_t)gm * N + gn] = c;
        } else if (mode == 1) {
          int part = gn >> 10;  // 0=q 1=k 2=v
          int d = gn & 1023;
          int b = gm >> 11, s = gm & 2047;
          if (part == 2) {
            vt[(size_t)gm * DD + d] = (_Float16)c;
          } else {
            int h = d >> 6, hd = d & 63;
            size_t sidx = (((size_t)(b * HH + h) * SS) + s) * HDI + hd;
            if (part == 0)
              qs[sidx] = (_Float16)(c * 0.125f);  // fold 1/sqrt(64)
            else
              ksp[sidx] = (_Float16)c;
          }
        } else {  // mode 2: head-transposed fused V: [b,h][hd][s]
          int b = gm >> 11, s = gm & 2047;
          int h = gn >> 6, hd = gn & 63;
          c *= latf[b * DD + gn];
          out_h[(((size_t)(b * HH + h) * HDI) + hd) * SS + s] = (_Float16)c;
        }
      }
    }
  }
}

// ---------------------------------------------------------------------------
// Flash attention (causal, MASKED_BIAS=-10000 on straddling tiles).
// Grid: B*H*(S/128) blocks, 256 threads (8 waves). Wave owns 16 query rows.
// K tiles [32 key][64 hd], V tiles [64 hd][32 key] (head-transposed source)
// streamed async into double-buffered LDS; per tile:
// 4 score WMMAs + online softmax + 4 PV WMMAs. Output merged-head f16.
// ---------------------------------------------------------------------------
#define KVPAD 72  // K tile row stride: 144B, 16B aligned
#define VPAD 40   // V tile row stride: 80B, 16B aligned
#define PPAD 40   // P scratch row stride: 80B, 16B aligned

__global__ __launch_bounds__(256) void k_flash(const _Float16* __restrict__ Q,
                                               const _Float16* __restrict__ Ks,
                                               const _Float16* __restrict__ Vst,
                                               _Float16* __restrict__ O) {
  __shared__ __align__(16) _Float16 Kl[2][32 * KVPAD];
  __shared__ __align__(16) _Float16 Vl[2][64 * VPAD];
  __shared__ __align__(16) _Float16 Pl[8 * 16 * PPAD];

  const int t = threadIdx.x;
  const int lane = t & 31, wave = t >> 5;
  const int bh = blockIdx.x >> 4;  // S/128 = 16 q-blocks per (b,h)
  const int qblk = blockIdx.x & 15;
  const int qwg = qblk * 128;
  const int qbase = qwg + wave * 16;
  const int hl = lane >> 4, n15 = lane & 15;

  const _Float16* Qp = Q + (size_t)bh * SS * HDI;
  const _Float16* Kp = Ks + (size_t)bh * SS * HDI;
  const _Float16* Vp = Vst + (size_t)bh * HDI * SS;  // [hd][s]
  _Float16* pw = Pl + wave * (16 * PPAD);

  // per-thread async-load slots (one b128 each)
  const int krow = t >> 3, kcg = (t & 7) * 8;  // K tile: 32 x 64
  const int vrow = t >> 2, vcg = (t & 3) * 8;  // V tile: 64 x 32

  // Q fragments (16x64 split into two K=32 chunks); Q pre-scaled by 0.125.
  v16h qf[2];
#pragma unroll
  for (int kc = 0; kc < 2; ++kc)
#pragma unroll
    for (int e = 0; e < 16; ++e)
      qf[kc][e] = Qp[(size_t)(qbase + n15) * HDI + kc * 32 + kpat(e, hl)];

  v8f o[4];
#pragma unroll
  for (int og = 0; og < 4; ++og) o[og] = zero8();
  float mi[8], li[8];
#pragma unroll
  for (int r = 0; r < 8; ++r) { mi[r] = -3.0e38f; li[r] = 0.0f; }

  const int nkb = (qwg + 128) >> 5;  // causal limit for this workgroup

  // prologue: stream key/value tile 0
  async_b128(lds_off(&Kl[0][krow * KVPAD + kcg]),
             &Kp[(size_t)krow * HDI + kcg]);
  async_b128(lds_off(&Vl[0][vrow * VPAD + vcg]), &Vp[(size_t)vrow * SS + vcg]);

  for (int kb = 0; kb < nkb; ++kb) {
    const int kbase = kb << 5;
    const int cur = kb & 1;
    wait_async0();
    __syncthreads();

    if (kb + 1 < nkb) {  // prefetch next K/V tile
      const int nb = (kb + 1) << 5;
      async_b128(lds_off(&Kl[cur ^ 1][krow * KVPAD + kcg]),
                 &Kp[(size_t)(nb + krow) * HDI + kcg]);
      async_b128(lds_off(&Vl[cur ^ 1][vrow * VPAD + vcg]),
                 &Vp[(size_t)vrow * SS + nb + vcg]);
    }

    if (kbase <= qbase + 15) {  // some row of this wave is unmasked
      const _Float16* Kc = Kl[cur];
      const _Float16* Vc = Vl[cur];
      v8f s0 = zero8(), s1 = zero8();
      v16h kf;
      // scores, keys 0..15 of tile
#pragma unroll
      for (int kc = 0; kc < 2; ++kc) {
#pragma unroll
        for (int e = 0; e < 16; ++e)
          kf[e] = Kc[n15 * KVPAD + kc * 32 + kpat(e, hl)];
        s0 = wmma16(qf[kc], kf, s0);
      }
      // scores, keys 16..31 of tile
#pragma unroll
      for (int kc = 0; kc < 2; ++kc) {
#pragma unroll
        for (int e = 0; e < 16; ++e)
          kf[e] = Kc[(16 + n15) * KVPAD + kc * 32 + kpat(e, hl)];
        s1 = wmma16(qf[kc], kf, s1);
      }

      // mask + online softmax (row stats reduced across 16-lane halves)
#pragma unroll
      for (int r = 0; r < 8; ++r) {
        int sq = qbase + r + 8 * hl;
        float v0 = s0[r];
        if (kbase + n15 > sq) v0 = -10000.0f;
        float v1 = s1[r];
        if (kbase + 16 + n15 > sq) v1 = -10000.0f;
        float rm = fmaxf(v0, v1);
#pragma unroll
        for (int d = 1; d < 16; d <<= 1) rm = fmaxf(rm, __shfl_xor(rm, d, 32));
        float mn = fmaxf(mi[r], rm);
        float alpha = __expf(mi[r] - mn);
        float p0 = __expf(v0 - mn);
        float p1 = __expf(v1 - mn);
        float rs = p0 + p1;
#pragma unroll
        for (int d = 1; d < 16; d <<= 1) rs += __shfl_xor(rs, d, 32);
        li[r] = li[r] * alpha + rs;
        mi[r] = mn;
#pragma unroll
        for (int og = 0; og < 4; ++og) o[og][r] *= alpha;
        int m = r + 8 * hl;
        pw[m * PPAD + n15] = (_Float16)p0;
        pw[m * PPAD + 16 + n15] = (_Float16)p1;
      }

      // P (16x32) -> A-fragment via per-wave LDS scratch, then PV
      v16h pf;
#pragma unroll
      for (int e = 0; e < 16; ++e) pf[e] = pw[n15 * PPAD + kpat(e, hl)];
#pragma unroll
      for (int og = 0; og < 4; ++og) {
        v16h vf;
#pragma unroll
        for (int e = 0; e < 16; ++e)
          vf[e] = Vc[(og * 16 + n15) * VPAD + kpat(e, hl)];
        o[og] = wmma16(pf, vf, o[og]);
      }
    }
  }

  // epilogue: normalize and write merged-head f16 [B*S, D]
  const int b = bh >> 4, h = bh & 15;
#pragma unroll
  for (int r = 0; r < 8; ++r) {
    float inv = 1.0f / li[r];
    int srow = qbase + r + 8 * hl;
    size_t base = ((size_t)(b * SS + srow)) * DD + h * HDI;
#pragma unroll
    for (int og = 0; og < 4; ++og)
      O[base + og * 16 + n15] = (_Float16)(o[og][r] * inv);
  }
}

// ---------------------------------------------------------------------------
extern "C" void kernel_launch(void* const* d_in, const int* in_sizes, int n_in,
                              void* d_out, int out_size, void* d_ws,
                              size_t ws_size, hipStream_t stream) {
  (void)in_sizes; (void)n_in; (void)out_size; (void)ws_size;
  const float* hidden   = (const float*)d_in[0];
  const float* latent   = (const float*)d_in[1];
  const float* c_attn_w = (const float*)d_in[2];
  const float* c_attn_b = (const float*)d_in[3];
  const float* c_proj_w = (const float*)d_in[4];
  const float* c_proj_b = (const float*)d_in[5];
  const float* text_w   = (const float*)d_in[6];
  const float* text_b   = (const float*)d_in[7];
  const float* lat_w    = (const float*)d_in[8];
  const float* lat_b    = (const float*)d_in[9];
  float* out = (float*)d_out;

  const size_t MT = (size_t)BB * SS;  // 8192 tokens
  char* p = (char*)d_ws;
  _Float16* Xh   = (_Float16*)p; p += MT * DD * 2;  // X f16; reused as attn-out
  _Float16* WAt  = (_Float16*)p; p += (size_t)3 * DD * DD * 2;  // [3D][D]
  _Float16* WTt  = (_Float16*)p; p += (size_t)DD * DD * 2;      // [D][D]
  _Float16* WPt  = (_Float16*)p; p += (size_t)DD * DD * 2;      // [D][D]
  float*    biasT = (float*)p;   p += DD * 4;
  float*    latf  = (float*)p;   p += (size_t)BB * DD * 4;
  _Float16* Qs   = (_Float16*)p; p += MT * DD * 2;
  _Float16* Ksp  = (_Float16*)p; p += MT * DD * 2;
  _Float16* Vt   = (_Float16*)p; p += MT * DD * 2;  // value stream, row-major
  _Float16* Vsp  = (_Float16*)p; p += MT * DD * 2;  // fused V, [b,h][hd][s]

  const int nX = (int)(MT * DD);
  k_cvt_f32_f16<<<(nX + 255) / 256, 256, 0, stream>>>(hidden, Xh, nX);
  // weights: transpose + convert
  k_cvt_t<<<dim3(3 * DD / 32, DD / 32), 256, 0, stream>>>(c_attn_w, WAt, DD,
                                                          3 * DD);
  k_cvt_t<<<dim3(DD / 32, DD / 32), 256, 0, stream>>>(c_proj_w, WPt, DD, DD);
  k_prep_text<<<(1025 * 1024 + 255) / 256, 256, 0, stream>>>(text_w, text_b,
                                                             WTt, biasT);
  k_latfus<<<(BB * DD + 255) / 256, 256, 0, stream>>>(latent, lat_w, lat_b,
                                                      latf);

  // GEMM1: QKV = X @ c_attn_w + b  -> split-head Q (scaled), K, row-major V
  k_gemm_f16<<<dim3(3 * DD / 128, (int)MT / 128), 256, 0, stream>>>(
      Xh, WAt, c_attn_b, (int)MT, 3 * DD, DD, 1, nullptr, nullptr, Qs, Ksp, Vt,
      nullptr);

  // GEMM2: LMF value fusion -> head-transposed fused V
  k_gemm_f16<<<dim3(DD / 128, (int)MT / 128), 256, 0, stream>>>(
      Vt, WTt, biasT, (int)MT, DD, DD, 2, nullptr, Vsp, nullptr, nullptr,
      nullptr, latf);

  // Flash attention -> merged-head f16 (reuses Xh buffer)
  k_flash<<<BB * HH * (SS / 128), 256, 0, stream>>>(Qs, Ksp, Vsp, Xh);

  // GEMM3: final projection -> f32 output
  k_gemm_f16<<<dim3(DD / 128, (int)MT / 128), 256, 0, stream>>>(
      Xh, WPt, c_proj_b, (int)MT, DD, DD, 0, out, nullptr, nullptr, nullptr,
      nullptr, nullptr);
}
